// LayerNormDenseGeneral_81570018885754
// MI455X (gfx1250) — matrix operand order repaired and checked by reading, not compiled, not run on hardware
//
#include <hip/hip_runtime.h>
#include <cstdint>
#include <cstddef>

// ---------------------------------------------------------------------------
// LayerNorm + Dense + LoRA fused as:
//   W_eff = kernel + lora_a @ lora_b          (tiny GEMM, stored transposed bf16)
//   y     = LayerNorm(x) in bf16
//   out   = y @ W_eff + bias                  (WMMA bf16 GEMM, f32 accumulate)
// MI455X: compute-bound GEMM -> v_wmma_f32_16x16x32_bf16 density is what counts.
// Wave tile 64x64 => 16 WMMA per 16 ds_load_b128 per K-slab (1.0 loads/WMMA).
// ---------------------------------------------------------------------------

typedef __bf16 bf16_t;
typedef __attribute__((ext_vector_type(16))) __bf16 v16bf;
typedef __attribute__((ext_vector_type(4)))  __bf16 v4bf;
typedef __attribute__((ext_vector_type(8)))  float  v8f;
typedef __attribute__((ext_vector_type(4)))  unsigned int u32x4;
typedef int v4i_gcc __attribute__((vector_size(16)));   // matches builtin param

#define BM 256          // block M tile
#define BN 128          // block N tile
#define BK 32           // K slab (one WMMA K)
#define LDT 40          // padded LDS row stride in bf16 elements (80B, bank-skewed)

// CDNA5 async global->LDS path (probe-verified to exist on this toolchain).
#if defined(__HIP_DEVICE_COMPILE__) && __has_builtin(__builtin_amdgcn_global_load_async_to_lds_b128)
#define USE_ASYNC_LDS 1
#else
#define USE_ASYNC_LDS 0
#endif

__device__ __forceinline__ void async_copy_b128(const bf16_t* g, bf16_t* l) {
#if USE_ASYNC_LDS
  __builtin_amdgcn_global_load_async_to_lds_b128(
      (__attribute__((address_space(1))) v4i_gcc*)(g),
      (__attribute__((address_space(3))) v4i_gcc*)(l),
      0, 0);
#else
  (void)g; (void)l;
#endif
}

__device__ __forceinline__ void wait_async_zero() {
#if USE_ASYNC_LDS
#if __has_builtin(__builtin_amdgcn_s_wait_asynccnt)
  __builtin_amdgcn_s_wait_asynccnt(0);
#else
  asm volatile("s_wait_asynccnt 0" ::: "memory");
#endif
#endif
}

// ---------------------------------------------------------------------------
// Kernel 1: W_eff^T (F x H, bf16) = (kernel + lora_a @ lora_b)^T
// Tiled with an LDS transpose so reads (row-major kernel) and writes
// (row-major W_eff^T) are both fully coalesced.
// ---------------------------------------------------------------------------
#define TTF 64
#define TTH 64
__global__ void __launch_bounds__(256)
build_wt_kernel(const float* __restrict__ W, const float* __restrict__ la,
                const float* __restrict__ lb, bf16_t* __restrict__ Wt,
                int H, int F, int R)
{
  const int f0 = blockIdx.x * TTF;
  const int h0 = blockIdx.y * TTH;
  __shared__ float lbs[32][TTF];
  __shared__ float tile[TTH][TTF + 1];

  const int Rc = (R < 32) ? R : 32;
  for (int idx = threadIdx.x; idx < Rc * TTF; idx += 256) {
    const int r = idx / TTF, ff = idx % TTF;
    lbs[r][ff] = lb[(size_t)r * F + f0 + ff];
  }
  __syncthreads();

  for (int idx = threadIdx.x; idx < TTH * TTF; idx += 256) {
    const int hh = idx / TTF, ff = idx % TTF;
    float sum = W[(size_t)(h0 + hh) * F + f0 + ff];
    const float* lar = la + (size_t)(h0 + hh) * R;
#pragma unroll 8
    for (int r = 0; r < Rc; ++r) sum += lar[r] * lbs[r][ff];
    tile[hh][ff] = sum;
  }
  __syncthreads();

  for (int idx = threadIdx.x; idx < TTH * TTF; idx += 256) {
    const int ff = idx / TTH, hh = idx % TTH;
    Wt[(size_t)(f0 + ff) * H + h0 + hh] = (bf16_t)tile[hh][ff];
  }
}

// ---------------------------------------------------------------------------
// Kernel 2: LayerNorm row -> bf16.  One block (8 wave32) per row.
// ---------------------------------------------------------------------------
__global__ void __launch_bounds__(256)
layernorm_bf16_kernel(const float* __restrict__ x, const float* __restrict__ gamma,
                      const float* __restrict__ beta, bf16_t* __restrict__ y, int H)
{
  const int row = blockIdx.x;
  const float* xr = x + (size_t)row * H;
  bf16_t* yr = y + (size_t)row * H;

  float s = 0.f, ss = 0.f;
  for (int i = threadIdx.x * 4; i < H; i += 256 * 4) {
    const float4 v = *(const float4*)(xr + i);
    s  += v.x + v.y + v.z + v.w;
    ss += v.x * v.x + v.y * v.y + v.z * v.z + v.w * v.w;
  }
#pragma unroll
  for (int off = 16; off > 0; off >>= 1) {      // wave32 butterfly
    s  += __shfl_xor(s,  off, 32);
    ss += __shfl_xor(ss, off, 32);
  }
  __shared__ float red[2][8];
  const int w = threadIdx.x >> 5;
  if ((threadIdx.x & 31) == 0) { red[0][w] = s; red[1][w] = ss; }
  __syncthreads();
  s = 0.f; ss = 0.f;
#pragma unroll
  for (int j = 0; j < 8; ++j) { s += red[0][j]; ss += red[1][j]; }

  const float invH = 1.0f / (float)H;
  const float mean = s * invH;
  const float var  = fmaxf(ss * invH - mean * mean, 0.f);
  const float rstd = rsqrtf(var + 1e-6f);

  for (int i = threadIdx.x * 4; i < H; i += 256 * 4) {
    const float4 v = *(const float4*)(xr + i);
    const float4 g = *(const float4*)(gamma + i);
    const float4 b = *(const float4*)(beta + i);
    v4bf o;
    o[0] = (bf16_t)((v.x - mean) * rstd * g.x + b.x);
    o[1] = (bf16_t)((v.y - mean) * rstd * g.y + b.y);
    o[2] = (bf16_t)((v.z - mean) * rstd * g.z + b.z);
    o[3] = (bf16_t)((v.w - mean) * rstd * g.w + b.w);
    *(v4bf*)(yr + i) = o;
  }
}

// ---------------------------------------------------------------------------
// Kernel 3: C(MxN,f32) = A(MxK,bf16) * Bt(NxK,bf16)^T + bias, WMMA bf16.
// Block = 256x128 tile, 8 waves in 4(M)x2(N); wave = 64x64 = 4x4 WMMA tiles.
// Double-buffered LDS (async global->LDS), padded stride, K stepped by 32.
//
// Fragment layouts per CDNA5 ISA 7.12.2 (wave32):
//  A 16x32 bf16 : lane m (0-15) holds K {0..7,16..23}; lane m+16 K {8..15,24..31}
//  B 32x16 bf16 : lane n (0-15) holds K {0..7,16..23} of col n; lane n+16 K {8..15,24..31}
//  C/D 16x16 f32: VGPR v, lanes 0-15 -> (M=v,   N=lane)
//                          lanes 16-31 -> (M=v+8, N=lane-16)
// ---------------------------------------------------------------------------
__device__ __forceinline__ v16bf load_frag_A(const bf16_t* tile, int row, int lane) {
  const int kh = (lane >> 4) << 3;              // 0 or 8
  const bf16_t* p = tile + row * LDT + kh;
  union { v16bf v; u32x4 u[2]; } f;
  f.u[0] = *(const u32x4*)(p);
  f.u[1] = *(const u32x4*)(p + 16);
  return f.v;
}

__device__ __forceinline__ v16bf load_frag_B(const bf16_t* tile, int row, int lane) {
  const int kb = (lane >> 4) << 3;              // 0 or 8 (K {kb..kb+7, kb+16..kb+23})
  const bf16_t* p = tile + row * LDT + kb;
  union { v16bf v; u32x4 u[2]; } f;
  f.u[0] = *(const u32x4*)(p);
  f.u[1] = *(const u32x4*)(p + 16);
  return f.v;
}

__global__ void __launch_bounds__(256)
wmma_gemm_bf16_kernel(const bf16_t* __restrict__ A, const bf16_t* __restrict__ Bt,
                      const float* __restrict__ bias, float* __restrict__ C,
                      int M, int N, int K)
{
  __shared__ bf16_t sA[2][BM * LDT];   // 2 x 20 KB
  __shared__ bf16_t sB[2][BN * LDT];   // 2 x 10 KB

  const int tid  = threadIdx.x;
  const int lane = tid & 31;
  const int wv   = tid >> 5;
  const int wm0  = (wv >> 1) * 64;   // wave M offset in block: 0/64/128/192
  const int wn0  = (wv & 1)  * 64;   // wave N offset in block: 0/64
  const int rowBase = blockIdx.y * BM;
  const int colBase = blockIdx.x * BN;

  // Staging: 16B chunks. A tile: 1024 chunks (4/thread); B tile: 512 (2/thread).
  const bf16_t* gA = A  + (size_t)rowBase * K;
  const bf16_t* gB = Bt + (size_t)colBase * K;

  int rA[4], kA[4];   // A chunk coords
#pragma unroll
  for (int t = 0; t < 4; ++t) {
    const int c = tid + t * 256;
    rA[t] = c >> 2;
    kA[t] = (c & 3) * 8;
  }
  int rB[2], kB[2];   // B chunk coords
#pragma unroll
  for (int t = 0; t < 2; ++t) {
    const int c = tid + t * 256;
    rB[t] = c >> 2;
    kB[t] = (c & 3) * 8;
  }

  v8f acc[4][4];
#pragma unroll
  for (int i = 0; i < 4; ++i)
#pragma unroll
    for (int j = 0; j < 4; ++j)
      acc[i][j] = (v8f){0.f, 0.f, 0.f, 0.f, 0.f, 0.f, 0.f, 0.f};

  const int nK = K / BK;

  // ---- prologue: stage K-slab 0 into buffer 0 ----
#if USE_ASYNC_LDS
#pragma unroll
  for (int t = 0; t < 4; ++t)
    async_copy_b128(gA + (size_t)rA[t] * K + kA[t], &sA[0][rA[t] * LDT + kA[t]]);
#pragma unroll
  for (int t = 0; t < 2; ++t)
    async_copy_b128(gB + (size_t)rB[t] * K + kB[t], &sB[0][rB[t] * LDT + kB[t]]);
  wait_async_zero();
#else
#pragma unroll
  for (int t = 0; t < 4; ++t)
    *(u32x4*)&sA[0][rA[t] * LDT + kA[t]] = *(const u32x4*)(gA + (size_t)rA[t] * K + kA[t]);
#pragma unroll
  for (int t = 0; t < 2; ++t)
    *(u32x4*)&sB[0][rB[t] * LDT + kB[t]] = *(const u32x4*)(gB + (size_t)rB[t] * K + kB[t]);
#endif
  __syncthreads();

  for (int kt = 0; kt < nK; ++kt) {
    const int cur = kt & 1;
    const int nxt = cur ^ 1;
    const size_t kOff = (size_t)(kt + 1) * BK;

#if !USE_ASYNC_LDS
    u32x4 stA[4], stB[2];
#endif
    if (kt + 1 < nK) {
#if USE_ASYNC_LDS
#pragma unroll
      for (int t = 0; t < 4; ++t)
        async_copy_b128(gA + (size_t)rA[t] * K + kOff + kA[t], &sA[nxt][rA[t] * LDT + kA[t]]);
#pragma unroll
      for (int t = 0; t < 2; ++t)
        async_copy_b128(gB + (size_t)rB[t] * K + kOff + kB[t], &sB[nxt][rB[t] * LDT + kB[t]]);
#else
#pragma unroll
      for (int t = 0; t < 4; ++t)
        stA[t] = *(const u32x4*)(gA + (size_t)rA[t] * K + kOff + kA[t]);
#pragma unroll
      for (int t = 0; t < 2; ++t)
        stB[t] = *(const u32x4*)(gB + (size_t)rB[t] * K + kOff + kB[t]);
#endif
    }

    // ---- compute on current buffer: 16 x v_wmma_f32_16x16x32_bf16 ----
    const bf16_t* tA = sA[cur];
    const bf16_t* tB = sB[cur];

    v16bf bfr[4];
#pragma unroll
    for (int j = 0; j < 4; ++j)
      bfr[j] = load_frag_B(tB, wn0 + j * 16 + (lane & 15), lane);

#pragma unroll
    for (int i = 0; i < 4; ++i) {
      const v16bf afr = load_frag_A(tA, wm0 + i * 16 + (lane & 15), lane);
#pragma unroll
      for (int j = 0; j < 4; ++j)
        acc[i][j] = __builtin_amdgcn_wmma_f32_16x16x32_bf16(
            false, afr, false, bfr[j], (short)0, acc[i][j], false, false);
    }

    if (kt + 1 < nK) {
#if USE_ASYNC_LDS
      wait_async_zero();
#else
#pragma unroll
      for (int t = 0; t < 4; ++t)
        *(u32x4*)&sA[nxt][rA[t] * LDT + kA[t]] = stA[t];
#pragma unroll
      for (int t = 0; t < 2; ++t)
        *(u32x4*)&sB[nxt][rB[t] * LDT + kB[t]] = stB[t];
#endif
    }
    __syncthreads();
  }

  // ---- epilogue: C/D layout -> global, + bias ----
#pragma unroll
  for (int i = 0; i < 4; ++i) {
    const int m = rowBase + wm0 + i * 16 + ((lane >> 4) * 8);
#pragma unroll
    for (int j = 0; j < 4; ++j) {
      const int n = colBase + wn0 + j * 16 + (lane & 15);
      const float bv = bias[n];
      float* cp = C + (size_t)m * N + n;
#pragma unroll
      for (int v = 0; v < 8; ++v)
        cp[(size_t)v * N] = acc[i][j][v] + bv;
    }
  }
}

// ---------------------------------------------------------------------------
// Launch
// ---------------------------------------------------------------------------
extern "C" void kernel_launch(void* const* d_in, const int* in_sizes, int n_in,
                              void* d_out, int out_size, void* d_ws, size_t ws_size,
                              hipStream_t stream) {
  const float* x       = (const float*)d_in[0];   // (B,S,H)
  const float* scale   = (const float*)d_in[1];   // (H)
  const float* ln_bias = (const float*)d_in[2];   // (H)
  const float* kernelW = (const float*)d_in[3];   // (H,F)
  const float* lora_a  = (const float*)d_in[4];   // (H,R)
  const float* lora_b  = (const float*)d_in[5];   // (R,F)
  const float* bias    = (const float*)d_in[6];   // (F)
  float* out = (float*)d_out;

  const int H = in_sizes[1];
  const int F = in_sizes[6];
  const int R = in_sizes[5] / F;
  const int M = (int)((long long)in_sizes[0] / H);   // B*S rows

  // Workspace: y (M*H bf16) then W_eff^T (F*H bf16)
  bf16_t* yb = (bf16_t*)d_ws;
  bf16_t* Wt = yb + (size_t)M * H;

  dim3 g1(F / TTF, H / TTH);
  build_wt_kernel<<<g1, 256, 0, stream>>>(kernelW, lora_a, lora_b, Wt, H, F, R);

  layernorm_bf16_kernel<<<M, 256, 0, stream>>>(x, scale, ln_bias, yb, H);

  dim3 g3(F / BN, M / BM);
  wmma_gemm_bf16_kernel<<<g3, 256, 0, stream>>>(yb, Wt, bias, out, M, F, H);
}
